// GraphAttentionLayer_24240795419407
// MI455X (gfx1250) — compile-verified, matrix operand order checked by reading, not run on hardware
//
#include <hip/hip_runtime.h>

// ---------------------------------------------------------------------------
// GAT dense layer for MI455X (gfx1250, wave32, WMMA).
//   h   = x @ W + b                        (WMMA bf16, f32 acc)
//   s1  = h@a1, s2 = h@a2, s2max = max(s2)
//   m_i = lrelu(s1_i + s2max + ab)         (exact row max; lrelu monotone)
//   out = [exp(lrelu(s1_i+s2_j+ab)-m_i)] @ h / Z_i   (fused, WMMA bf16)
// The 8192x8192 attention matrix is never materialized (would be 256 MB of
// HBM traffic each way at 23.3 TB/s; instead logits are regenerated from the
// rank-1 structure in registers). Dominant cost: 34.4 GFLOP of bf16 WMMA.
// ---------------------------------------------------------------------------

typedef __bf16 v16bf __attribute__((ext_vector_type(16)));
typedef __bf16 v8bf  __attribute__((ext_vector_type(8)));
typedef float  v8f   __attribute__((ext_vector_type(8)));
typedef float  v4f   __attribute__((ext_vector_type(4)));

#define NEG_SLOPE 0.01f

static __device__ __forceinline__ v8f wmma_bf16(v16bf a, v16bf b, v8f c) {
  // D(16x16,f32) = A(16x32,bf16) * B(32x16,bf16) + C
  return __builtin_amdgcn_wmma_f32_16x16x32_bf16(
      /*neg_a=*/false, a, /*neg_b=*/false, b,
      /*c_mod=*/(short)0, c, /*reuse_a=*/false, /*reuse_b=*/false);
}

// B-operand packed layout: fragment (k0 mult of 32, n0 mult of 16) occupies
// 512 bf16; lane L holds 16 contiguous bf16 at frag*512 + L*16.
//   element (k,n): lane = (n&15) + bit3(k)*16 ; e = (k&7) + bit4(k)*8
static __device__ __forceinline__ size_t b_pack_idx(int k, int n, int D) {
  int fragI = (k >> 5) * (D >> 4) + (n >> 4);
  int lane  = (n & 15) + (((k >> 3) & 1) << 4);
  int e     = (k & 7) + (((k >> 4) & 1) << 3);
  return (size_t)fragI * 512 + lane * 16 + e;
}

// --- pack W_fc (f32 row-major KxD) into bf16 B-fragment layout -------------
__global__ void k_pack_w(const float* __restrict__ W, __bf16* __restrict__ Wp,
                         int K, int D) {
  int idx = blockIdx.x * blockDim.x + threadIdx.x;
  if (idx >= K * D) return;
  int k = idx / D, n = idx - k * D;
  Wp[b_pack_idx(k, n, D)] = (__bf16)W[idx];
}

// --- h = x @ W + b ; also emit h pre-packed as B operand of the attn GEMM --
// block = 256 thr (8 waves); block tile 128(M) x 64(N); wave tile 32x32.
__global__ __launch_bounds__(256) void k_fc_gemm(
    const float* __restrict__ x, const __bf16* __restrict__ Wp,
    const float* __restrict__ bfc, float* __restrict__ h,
    __bf16* __restrict__ hp, int N, int K, int D) {
  const int t = threadIdx.x;
  const int lane = t & 31, w = t >> 5;
  const int m0 = blockIdx.x * 128 + (w >> 1) * 32;
  const int n0 = blockIdx.y * 64 + (w & 1) * 32;
  const int mrow = m0 + (lane & 15);     // A-frag row for this lane
  const int abase = (lane >> 4) * 8;     // K-half select per A-frag layout
  const v16bf* Wpv = (const v16bf*)Wp;

  v8f acc[2][2] = {};
  for (int k0 = 0; k0 < K; k0 += 32) {
    v16bf a[2];
#pragma unroll
    for (int mt = 0; mt < 2; ++mt) {
      const float* xp = x + (size_t)(mrow + mt * 16) * K + k0 + abase;
      v4f f0 = *(const v4f*)(xp);
      v4f f1 = *(const v4f*)(xp + 4);
      v4f f2 = *(const v4f*)(xp + 16);
      v4f f3 = *(const v4f*)(xp + 20);
#pragma unroll
      for (int i = 0; i < 4; ++i) {
        a[mt][i]      = (__bf16)f0[i];
        a[mt][4 + i]  = (__bf16)f1[i];
        a[mt][8 + i]  = (__bf16)f2[i];
        a[mt][12 + i] = (__bf16)f3[i];
      }
    }
    v16bf b[2];
#pragma unroll
    for (int nt = 0; nt < 2; ++nt) {
      int fragI = (k0 >> 5) * (D >> 4) + ((n0 + nt * 16) >> 4);
      b[nt] = Wpv[(size_t)fragI * 32 + lane];   // 32B contiguous per lane
    }
#pragma unroll
    for (int mt = 0; mt < 2; ++mt)
#pragma unroll
      for (int nt = 0; nt < 2; ++nt)
        acc[mt][nt] = wmma_bf16(a[mt], b[nt], acc[mt][nt]);
  }
  // epilogue: +bias, store h (f32) and packed-B bf16 copy for the attn GEMM
#pragma unroll
  for (int mt = 0; mt < 2; ++mt) {
#pragma unroll
    for (int nt = 0; nt < 2; ++nt) {
      int n = n0 + nt * 16 + (lane & 15);
      float bias = bfc[n];
#pragma unroll
      for (int r = 0; r < 8; ++r) {
        int m = m0 + mt * 16 + (lane >> 4) * 8 + r;   // C-frag row
        float v = acc[mt][nt][r] + bias;
        h[(size_t)m * D + n] = v;
        hp[b_pack_idx(m, n, D)] = (__bf16)v;          // m is K-dim of GEMM2
      }
    }
  }
}

// --- s1/s2 GEMVs + per-block max of s2 -------------------------------------
__global__ __launch_bounds__(256) void k_scores(
    const float* __restrict__ h, const float* __restrict__ a1,
    const float* __restrict__ a2, float* __restrict__ s1,
    float* __restrict__ s2, float* __restrict__ bmax, int N, int D) {
  __shared__ float sm[8];
  const int lane = threadIdx.x & 31, w = threadIdx.x >> 5;
  const int row = blockIdx.x * 8 + w;
  const float* hr = h + (size_t)row * D;
  float p1 = 0.f, p2 = 0.f;
  for (int e = lane; e < D; e += 32) {
    float hv = hr[e];
    p1 += hv * a1[e];
    p2 += hv * a2[e];
  }
#pragma unroll
  for (int off = 16; off > 0; off >>= 1) {
    p1 += __shfl_xor(p1, off, 32);
    p2 += __shfl_xor(p2, off, 32);
  }
  if (lane == 0) { s1[row] = p1; s2[row] = p2; sm[w] = p2; }
  __syncthreads();
  if (threadIdx.x == 0) {
    float mx = sm[0];
#pragma unroll
    for (int i = 1; i < 8; ++i) mx = fmaxf(mx, sm[i]);
    bmax[blockIdx.x] = mx;
  }
}

__global__ __launch_bounds__(256) void k_max_reduce(
    const float* __restrict__ bmax, float* __restrict__ s2mx, int nb) {
  __shared__ float sm[256];
  float mx = -3.4e38f;
  for (int i = threadIdx.x; i < nb; i += 256) mx = fmaxf(mx, bmax[i]);
  sm[threadIdx.x] = mx;
  __syncthreads();
  for (int s = 128; s > 0; s >>= 1) {
    if (threadIdx.x < (unsigned)s)
      sm[threadIdx.x] = fmaxf(sm[threadIdx.x], sm[threadIdx.x + s]);
    __syncthreads();
  }
  if (threadIdx.x == 0) s2mx[0] = sm[0];
}

// --- fused softmax-attention GEMM: out = softmax(lrelu(s1_i+s2_j+ab)) @ h --
// block = 256 thr (8 waves); block tile 64(M rows i) x 256(full feature N);
// K loop over j in steps of 64. Per step, the 256 threads cooperatively
// build the 64x64 P tile (exp computed once), staged in LDS in A-fragment
// layout; each wave owns a 64x32 output slab (4 M-tiles x 2 N-tiles).
__global__ __launch_bounds__(256) void k_attn_gemm(
    const __bf16* __restrict__ hp, const float* __restrict__ s1,
    const float* __restrict__ s2, const float* __restrict__ s2maxp,
    const float* __restrict__ abp, float* __restrict__ out, int N, int D) {
  __shared__ __bf16 pLds[8 * 512];   // 8 A-frags (16x32) of the 64x64 P tile
  __shared__ float  zp[64 * 4];      // per-(row,quarter) Z partials
  __shared__ float  zr[64];          // 1/Z per row
  const int t = threadIdx.x;
  const int lane = t & 31, w = t >> 5;
  const int m0 = blockIdx.x * 64;
  const float ab = abp[0];
  const float s2mx = s2maxp[0];

  // P-producer mapping: thread t -> local row rr (4 threads/row), quarter q
  const int rr = t >> 2, q = t & 3;
  const float s1r = s1[m0 + rr];
  float mrow = s1r + s2mx + ab;                 // exact row max of logits
  mrow = mrow > 0.f ? mrow : mrow * NEG_SLOPE;
  const int msubW = rr >> 4, mm = rr & 15, kchunk = q >> 1;
  const int esel = (q & 1) * 8;                 // which e-half of the frag
  const int offA = (msubW * 2 + kchunk) * 512 + mm * 16 + esel;        // i=0..7
  const int offB = (msubW * 2 + kchunk) * 512 + (mm + 16) * 16 + esel; // i=8..15

  const int n0w = w * 32;                       // wave's 32 feature columns
  const v16bf* hpv = (const v16bf*)hp;

  v8f acc[4][2] = {};
  float zpart = 0.f;

  for (int k0 = 0; k0 < N; k0 += 64) {
    __syncthreads();   // previous iteration's A-frag reads done
    {
      const float* s2p = s2 + k0 + q * 16;
      v8bf pa, pb;
#pragma unroll
      for (int i = 0; i < 8; ++i) {
        float e0 = s1r + s2p[i] + ab;
        e0 = e0 > 0.f ? e0 : e0 * NEG_SLOPE;
        float p0 = __expf(e0 - mrow);
        zpart += p0; pa[i] = (__bf16)p0;
        float e1 = s1r + s2p[i + 8] + ab;
        e1 = e1 > 0.f ? e1 : e1 * NEG_SLOPE;
        float p1 = __expf(e1 - mrow);
        zpart += p1; pb[i] = (__bf16)p1;
      }
      *(v8bf*)&pLds[offA] = pa;   // ds_store_b128
      *(v8bf*)&pLds[offB] = pb;   // ds_store_b128
    }
    __syncthreads();
#pragma unroll
    for (int kc = 0; kc < 2; ++kc) {
      int fb = ((k0 >> 5) + kc) * (D >> 4) + (n0w >> 4);
      v16bf b0 = hpv[(size_t)fb * 32 + lane];
      v16bf b1 = hpv[(size_t)(fb + 1) * 32 + lane];
#pragma unroll
      for (int ms = 0; ms < 4; ++ms) {
        v16bf a = *(const v16bf*)&pLds[(ms * 2 + kc) * 512 + lane * 16];
        acc[ms][0] = wmma_bf16(a, b0, acc[ms][0]);
        acc[ms][1] = wmma_bf16(a, b1, acc[ms][1]);
      }
    }
  }

  // deterministic Z reduction (4 partials per row), then normalize + store
  zp[rr * 4 + q] = zpart;
  __syncthreads();
  if (t < 64)
    zr[t] = 1.0f / (zp[t * 4] + zp[t * 4 + 1] + zp[t * 4 + 2] + zp[t * 4 + 3]);
  __syncthreads();
#pragma unroll
  for (int ms = 0; ms < 4; ++ms) {
#pragma unroll
    for (int nt = 0; nt < 2; ++nt) {
      int n = n0w + nt * 16 + (lane & 15);
#pragma unroll
      for (int r = 0; r < 8; ++r) {
        int ml = ms * 16 + (lane >> 4) * 8 + r;    // C-frag row
        out[(size_t)(m0 + ml) * D + n] = acc[ms][nt][r] * zr[ml];
      }
    }
  }
}

extern "C" void kernel_launch(void* const* d_in, const int* in_sizes, int n_in,
                              void* d_out, int out_size, void* d_ws, size_t ws_size,
                              hipStream_t stream) {
  (void)n_in; (void)out_size; (void)ws_size;
  const float* x   = (const float*)d_in[0];
  const float* Wfc = (const float*)d_in[1];
  const float* bfc = (const float*)d_in[2];
  const float* a1  = (const float*)d_in[3];
  const float* a2  = (const float*)d_in[4];
  const float* abp = (const float*)d_in[5];

  const int D = in_sizes[2];            // out_dim (256)
  const int K = in_sizes[1] / D;        // in_dim  (256)
  const int N = in_sizes[0] / K;        // rows    (8192)

  // workspace carve (~12.3 MB)
  char* ws = (char*)d_ws;
  size_t off = 0;
  auto carve = [&](size_t bytes) -> void* {
    void* p = ws + off;
    off += (bytes + 255) & ~(size_t)255;
    return p;
  };
  float*  h    = (float*) carve((size_t)N * D * sizeof(float));
  __bf16* hp   = (__bf16*)carve((size_t)N * D * sizeof(__bf16));
  __bf16* Wp   = (__bf16*)carve((size_t)K * D * sizeof(__bf16));
  float*  s1   = (float*) carve((size_t)N * sizeof(float));
  float*  s2   = (float*) carve((size_t)N * sizeof(float));
  const int sBlocks = N / 8;
  float*  bmax = (float*) carve((size_t)sBlocks * sizeof(float));
  float*  s2mx = (float*) carve(256);

  k_pack_w<<<(K * D + 255) / 256, 256, 0, stream>>>(Wfc, Wp, K, D);
  k_fc_gemm<<<dim3(N / 128, D / 64), 256, 0, stream>>>(x, Wp, bfc, h, hp, N, K, D);
  k_scores<<<sBlocks, 256, 0, stream>>>(h, a1, a2, s1, s2, bmax, N, D);
  k_max_reduce<<<1, 256, 0, stream>>>(bmax, s2mx, sBlocks);
  k_attn_gemm<<<N / 64, 256, 0, stream>>>(hp, s1, s2, s2mx, abp, (float*)d_out, N, D);
}
